// Linear_19920058319103
// MI455X (gfx1250) — matrix operand order repaired and checked by reading, not compiled, not run on hardware
//
#include <hip/hip_runtime.h>
#include <hip/hip_bf16.h>

typedef __attribute__((ext_vector_type(16))) _Float16 v16h;
typedef __attribute__((ext_vector_type(8)))  _Float16 v8h;
typedef __attribute__((ext_vector_type(8)))  float    v8f;
typedef __attribute__((ext_vector_type(4)))  float    v4f;
typedef __attribute__((ext_vector_type(4)))  _Float16 v4h;

#define BN    16384      // B*N = 4*4096 rows
#define DD    1024       // feature dim
#define RR    8          // TT rank
#define NCOL  144        // packed columns: 8 (c0) + 64 (c1) + 64 (c2) + 8 pad
#define NTIL  9          // 144/16 column tiles
#define KCH   32         // 1024/32 K-chunks

// ---------------- prep: f32 -> f16 activation convert (vectorized) ----------
__global__ void tt_cvt_f32_f16(const float* __restrict__ s,
                               _Float16* __restrict__ d, int n4) {
  for (int i = blockIdx.x * blockDim.x + threadIdx.x; i < n4;
       i += gridDim.x * blockDim.x) {
    v4f f = ((const v4f*)s)[i];
    v4h h;
    h[0] = (_Float16)f[0]; h[1] = (_Float16)f[1];
    h[2] = (_Float16)f[2]; h[3] = (_Float16)f[3];
    ((v4h*)d)[i] = h;
  }
}

// -------- prep: pack core0|core1|core2 into WMMA-B fragment order -----------
// bfrag[kc][t][lane][e] : column = t*16 + (lane&15), K = kc*32 + (lane>>4)*16 + e
__global__ void tt_pack_b(const float* __restrict__ c0,   // (1,1024,8)
                          const float* __restrict__ c1,   // (8,1024,8)
                          const float* __restrict__ c2,   // (8,1024,8)
                          _Float16* __restrict__ bfrag) {
  int idx = blockIdx.x * blockDim.x + threadIdx.x;        // over kc*t*lane
  if (idx >= KCH * NTIL * 32) return;
  int lane = idx & 31;
  int t    = (idx >> 5) % NTIL;
  int kc   = idx / (NTIL * 32);
  int col  = t * 16 + (lane & 15);
  int klo  = (lane >> 4) * 16;
  _Float16* dst = bfrag + (size_t)idx * 16;
#pragma unroll
  for (int e = 0; e < 16; ++e) {
    int K = kc * 32 + klo + e;
    float v = 0.f;
    if (col < 8)        v = c0[(size_t)K * 8 + col];
    else if (col < 72)  { int c = col - 8;  v = c1[(size_t)(c >> 3) * 8192 + (size_t)K * 8 + (c & 7)]; }
    else if (col < 136) { int c = col - 72; v = c2[(size_t)(c >> 3) * 8192 + (size_t)K * 8 + (c & 7)]; }
    dst[e] = (_Float16)v;
  }
}

// -------- prep: pack core3 (8x1024) into WMMA-B fragments, K padded to 32 ---
// b3frag[t2][lane][e] : column = t2*16 + (lane&15), K = (lane>>4)*16 + e (valid K<8)
__global__ void tt_pack_b3(const float* __restrict__ c3,  // (8,1024,1)
                           _Float16* __restrict__ b3frag) {
  int idx = blockIdx.x * blockDim.x + threadIdx.x;        // over 64*32
  if (idx >= 64 * 32) return;
  int lane = idx & 31;
  int t2   = idx >> 5;
  int col  = t2 * 16 + (lane & 15);
  int klo  = (lane >> 4) * 16;
  _Float16* dst = b3frag + (size_t)idx * 16;
#pragma unroll
  for (int e = 0; e < 16; ++e) {
    int K = klo + e;
    dst[e] = (_Float16)((K < RR) ? c3[(size_t)K * DD + col] : 0.f);
  }
}

// ---------------------------- fused TT depth step ---------------------------
// One wave per 16-row tile; 4 waves per block; grid = BN/64 = 256 blocks.
template <bool FINAL>
__global__ void __launch_bounds__(128)
tt_step(const _Float16* __restrict__ xh,      // BN x 1024 f16 activations
        const _Float16* __restrict__ bfrag,   // packed c0|c1|c2 fragments
        const _Float16* __restrict__ b3frag,  // packed c3 fragments
        const float* __restrict__ bias,
        _Float16* __restrict__ xh_out,        // step-1 output (f16)
        float* __restrict__ fout) {           // step-2 output (f32)
  __shared__ float Tsh[4][16][NCOL + 4];      // per-wave T tile (pad vs banks)
  __shared__ float Wsh[4][16][RR];            // per-wave w2

  const int lane = threadIdx.x & 31;
  const int wv   = threadIdx.x >> 5;
  const int tile = blockIdx.x * 4 + wv;       // 16-row tile id, 0..1023
  const int row0 = tile * 16;
  const int m    = lane & 15;                 // A row / D column lane-group
  const int kb   = (lane >> 4) * 16;          // A-fragment K sub-base

  const _Float16* xrow = xh + (size_t)(row0 + m) * DD + kb;

  // ---- fused triple GEMM: acc[t] = X_tile @ Bpack[:, t*16:t*16+16] ----
  v8f acc[NTIL];
#pragma unroll
  for (int t = 0; t < NTIL; ++t)
#pragma unroll
    for (int j = 0; j < 8; ++j) acc[t][j] = 0.f;

  for (int kc = 0; kc < KCH; ++kc) {
    union { v16h v; v8h h[2]; } a;
    const _Float16* ap = xrow + kc * 32;
    a.h[0] = *(const v8h*)(ap);               // K = kc*32 + kb   .. +7
    a.h[1] = *(const v8h*)(ap + 8);           // K = kc*32 +8+kb  .. +7
    if (kc + 4 < KCH) __builtin_prefetch(xrow + (kc + 4) * 32, 0, 1);
    const _Float16* bp = bfrag + ((size_t)(kc * NTIL) * 32 + lane) * 16;
#pragma unroll
    for (int t = 0; t < NTIL; ++t) {
      v16h b = *(const v16h*)(bp + (size_t)t * 32 * 16);
      acc[t] = __builtin_amdgcn_wmma_f32_16x16x32_f16(
          false, a.v, false, b, (short)0, acc[t], false, false);
    }
  }

  // ---- spill T tile to LDS (C layout: M = j + 8*(lane>>4), N = lane&15) ----
#pragma unroll
  for (int t = 0; t < NTIL; ++t)
#pragma unroll
    for (int j = 0; j < 8; ++j)
      Tsh[wv][j + 8 * (lane >> 4)][t * 16 + (lane & 15)] = acc[t][j];
  __syncthreads();

  // ---- tiny rank contractions: w1 = w0.T1, w2 = w1.T2 (one row per lane) ---
  if (lane < 16) {
    const float* Trow = &Tsh[wv][lane][0];
    float w1[RR], w2[RR];
#pragma unroll
    for (int R = 0; R < RR; ++R) {
      float s = 0.f;
#pragma unroll
      for (int r = 0; r < RR; ++r) s += Trow[r] * Trow[8 + r * 8 + R];
      w1[R] = s;
    }
#pragma unroll
    for (int R = 0; R < RR; ++R) {
      float s = 0.f;
#pragma unroll
      for (int r = 0; r < RR; ++r) s += w1[r] * Trow[72 + r * 8 + R];
      w2[R] = s;
      Wsh[wv][lane][R] = s;
    }
  }
  __syncthreads();

  // ---- output GEMM: X' = w2(16x8, K-padded to 32) @ C3(8x1024) -------------
  union { v16h v; v8h h[2]; } a2;
#pragma unroll
  for (int e = 0; e < 16; ++e) {
    int K = kb + e;
    a2.v[e] = (_Float16)((K < RR) ? Wsh[wv][m][K] : 0.f);
  }
  for (int t2 = 0; t2 < 64; ++t2) {
    v16h b3 = *(const v16h*)(b3frag + ((size_t)(t2 * 32) + lane) * 16);
    v8f d;
#pragma unroll
    for (int j = 0; j < 8; ++j) d[j] = 0.f;
    d = __builtin_amdgcn_wmma_f32_16x16x32_f16(
        false, a2.v, false, b3, (short)0, d, false, false);
    int col = t2 * 16 + (lane & 15);
#pragma unroll
    for (int j = 0; j < 8; ++j) {
      int row = row0 + j + 8 * (lane >> 4);
      if (FINAL) fout[(size_t)row * DD + col] = d[j] + bias[col];
      else       xh_out[(size_t)row * DD + col] = (_Float16)d[j];
    }
  }
}

// ------------------------------- launcher -----------------------------------
extern "C" void kernel_launch(void* const* d_in, const int* in_sizes, int n_in,
                              void* d_out, int out_size, void* d_ws, size_t ws_size,
                              hipStream_t stream) {
  const float* x    = (const float*)d_in[0];
  const float* c0   = (const float*)d_in[1];
  const float* c1   = (const float*)d_in[2];
  const float* c2   = (const float*)d_in[3];
  const float* c3   = (const float*)d_in[4];
  const float* bias = (const float*)d_in[5];
  float* out = (float*)d_out;

  // workspace layout (256B aligned)
  char* ws = (char*)d_ws;
  size_t off = 0;
  auto take = [&](size_t bytes) {
    size_t o = off;
    off = (off + bytes + 255) & ~(size_t)255;
    return o;
  };
  _Float16* bfrag  = (_Float16*)(ws + take((size_t)KCH * NTIL * 32 * 16 * 2));
  _Float16* b3frag = (_Float16*)(ws + take((size_t)64 * 32 * 16 * 2));
  _Float16* xhA    = (_Float16*)(ws + take((size_t)BN * DD * 2));
  _Float16* xhB    = (_Float16*)(ws + take((size_t)BN * DD * 2));
  (void)ws_size; (void)in_sizes; (void)n_in; (void)out_size;

  // prep
  tt_cvt_f32_f16<<<4096, 256, 0, stream>>>(x, xhA, BN * DD / 4);
  tt_pack_b<<<(KCH * NTIL * 32 + 255) / 256, 256, 0, stream>>>(c0, c1, c2, bfrag);
  tt_pack_b3<<<(64 * 32 + 255) / 256, 256, 0, stream>>>(c3, b3frag);

  // depth = 2 fused steps
  tt_step<false><<<BN / 64, 128, 0, stream>>>(xhA, bfrag, b3frag, bias, xhB, nullptr);
  tt_step<true ><<<BN / 64, 128, 0, stream>>>(xhB, bfrag, b3frag, bias, nullptr, out);
}
